// Encoder_42236708389388
// MI455X (gfx1250) — matrix operand order repaired and checked by reading, not compiled, not run on hardware
//
#include <hip/hip_runtime.h>
#include <stddef.h>

// Problem constants (match reference)
#define DIM_I 1024
#define DIM_H 2048
#define DIM_O 1024
#define DIM_T 8192
#define NB    64          // persistent blocks for the GRU scan
#define JB    (DIM_H/NB)  // h-elements per block = 32

typedef __attribute__((ext_vector_type(16))) __bf16 v16bf;
typedef __attribute__((ext_vector_type(8)))  __bf16 v8bf;
typedef __attribute__((ext_vector_type(8)))  float  v8f;

// ---------------------------------------------------------------------------
// f32 -> bf16 conversion (grid-stride)
// ---------------------------------------------------------------------------
__global__ void cvt_f32_bf16(const float* __restrict__ src,
                             __bf16* __restrict__ dst, size_t n) {
    size_t i = (size_t)blockIdx.x * blockDim.x + threadIdx.x;
    size_t stride = (size_t)gridDim.x * blockDim.x;
    for (; i < n; i += stride) dst[i] = (__bf16)src[i];
}

// ---------------------------------------------------------------------------
// bf16 WMMA GEMM:  C[M,N] = A[M,K] * B[N,K]^T + bias[N]   (f32 accumulate)
// Register-blocked: each wave computes a 64x64 output block as a 4x4 grid of
// 16x16 WMMA tiles. Per K-step of 32: 16 B128 loads feed 16 v_wmma ops
// (1 load per wmma; A/B fragments each reused 4x in registers).
// Operand layouts per CDNA5 ISA 7.12.2 (16-bit A 16x32, B 32x16).
// Requires M % 64 == 0, N % 64 == 0, K % 32 == 0.
// ---------------------------------------------------------------------------
__global__ void __launch_bounds__(256)
gemm_bf16_wmma(const __bf16* __restrict__ A, const __bf16* __restrict__ B,
               const float* __restrict__ bias, float* __restrict__ C,
               int M, int N, int K) {
    const int lane = threadIdx.x & 31;
    const int wid  = blockIdx.x * 8 + (threadIdx.x >> 5);   // 8 waves / block
    const int nstrips = N >> 6;                             // 64-wide strips
    const int ms = wid / nstrips;
    const int ns = wid % nstrips;
    if (ms >= (M >> 6)) return;          // wave-uniform: EXEC stays all-ones

    const int half = lane >> 4;          // 0: lanes 0-15, 1: lanes 16-31
    const int l    = lane & 15;

    const __bf16* arow0 = A + (size_t)(ms * 64 + l) * K;   // A rows, +16*i per tile
    const __bf16* brow0 = B + (size_t)(ns * 64 + l) * K;   // weight rows, +16*j

    union Pack { v16bf v; v8bf h[2]; };

    v8f acc[4][4];
#pragma unroll
    for (int i = 0; i < 4; ++i)
#pragma unroll
        for (int j = 0; j < 4; ++j) acc[i][j] = (v8f){};

    for (int k0 = 0; k0 < K; k0 += 32) {
        Pack a[4], b[4];
#pragma unroll
        for (int i = 0; i < 4; ++i) {
            const __bf16* ap = arow0 + (size_t)(i * 16) * K + k0;
            // A lanes 0-15: K = {0..7, 16..23}; lanes 16-31: K = {8..15, 24..31}
            a[i].h[0] = *(const v8bf*)(ap + half * 8);
            a[i].h[1] = *(const v8bf*)(ap + 16 + half * 8);
        }
#pragma unroll
        for (int j = 0; j < 4; ++j) {
            const __bf16* bp = brow0 + (size_t)(j * 16) * K + k0;
            // B lanes 0-15: K = {0..7, 16..23}; lanes 16-31: K = {8..15, 24..31}  (N = lane&15)
            b[j].h[0] = *(const v8bf*)(bp + half * 8);
            b[j].h[1] = *(const v8bf*)(bp + 16 + half * 8);
        }
#pragma unroll
        for (int i = 0; i < 4; ++i)
#pragma unroll
            for (int j = 0; j < 4; ++j)
                acc[i][j] = __builtin_amdgcn_wmma_f32_16x16x32_bf16(
                    false, a[i].v, false, b[j].v, (short)0, acc[i][j],
                    false, false);
    }

    // C/D layout: VGPR v -> M = m0 + v + half*8, N = n0 + (lane&15)
#pragma unroll
    for (int j = 0; j < 4; ++j) {
        const int col = ns * 64 + j * 16 + l;
        const float bc = bias ? bias[col] : 0.0f;
#pragma unroll
        for (int i = 0; i < 4; ++i) {
#pragma unroll
            for (int v = 0; v < 8; ++v) {
                int row = ms * 64 + i * 16 + v + half * 8;
                C[(size_t)row * N + col] = acc[i][j][v] + bc;
            }
        }
    }
}

// ---------------------------------------------------------------------------
// init: zero h state and grid-barrier counter (stream-ordered, replay-safe)
// ---------------------------------------------------------------------------
__global__ void init_state(float* __restrict__ h, unsigned int* __restrict__ cnt) {
    for (int i = threadIdx.x; i < DIM_H; i += blockDim.x) h[i] = 0.0f;
    if (threadIdx.x == 0)
        __hip_atomic_store(cnt, 0u, __ATOMIC_RELAXED, __HIP_MEMORY_SCOPE_AGENT);
}

// ---------------------------------------------------------------------------
// Persistent GRU scan. 64 blocks; block b owns h[j0 .. j0+31] and the three
// w_hh rows (r/z/n) per element. w_hh (48 MB f32) stays resident in the
// 192 MB L2; each step streams the block's 96 rows from L2.
// Grid-wide barrier per step via L2 atomic counter (release/acquire).
// ---------------------------------------------------------------------------
__global__ void __launch_bounds__(256)
gru_scan(const float* __restrict__ whh, const float* __restrict__ igates,
         const float* __restrict__ b_n, float* __restrict__ h,
         __bf16* __restrict__ hs, unsigned int* __restrict__ cnt) {
    __shared__ float hsh[DIM_H];
    __shared__ float acc[3 * JB];

    const int tid  = threadIdx.x;
    const int lane = tid & 31;
    const int wave = tid >> 5;           // 8 waves
    const int j0   = blockIdx.x * JB;

    for (int t = 0; t < DIM_T; ++t) {
        // Stage full h into LDS
        for (int i = tid; i < DIM_H; i += 256) hsh[i] = h[i];
        __syncthreads();

        // 96 rows (3 gates x 32 elements), 12 per wave; 32-lane dot products
        for (int p = wave; p < 3 * JB; p += 8) {
            const int g = p / JB;
            const int j = p % JB;
            const float* row = whh + ((size_t)(g * DIM_H + j0 + j)) * DIM_H;
            float s = 0.0f;
#pragma unroll 8
            for (int k = lane; k < DIM_H; k += 32)
                s = fmaf(row[k], hsh[k], s);
            s += __shfl_xor(s, 16);
            s += __shfl_xor(s, 8);
            s += __shfl_xor(s, 4);
            s += __shfl_xor(s, 2);
            s += __shfl_xor(s, 1);
            if (lane == 0) acc[p] = s;
        }
        __syncthreads();

        // Gate nonlinearity + state update for this block's 32 elements
        if (tid < JB) {
            const int j = j0 + tid;
            const float* ig = igates + (size_t)t * (3 * DIM_H);
            float r = 1.0f / (1.0f + expf(-(ig[j]             + acc[tid])));
            float z = 1.0f / (1.0f + expf(-(ig[DIM_H + j]     + acc[JB + tid])));
            float n = tanhf(ig[2 * DIM_H + j] + r * (acc[2 * JB + tid] + b_n[j]));
            float hn = n + z * (hsh[j] - n);
            h[j] = hn;
            hs[(size_t)t * DIM_H + j] = (__bf16)hn;
        }

        // -------- grid-wide barrier --------
        __threadfence();
        __syncthreads();
        if (tid == 0) {
            __hip_atomic_fetch_add(cnt, 1u, __ATOMIC_RELEASE,
                                   __HIP_MEMORY_SCOPE_AGENT);
            const unsigned target = (unsigned)NB * (unsigned)(t + 1);
            while (__hip_atomic_load(cnt, __ATOMIC_ACQUIRE,
                                     __HIP_MEMORY_SCOPE_AGENT) < target)
                __builtin_amdgcn_s_sleep(1);
        }
        __syncthreads();
        // per-wave acquire so every CU's L0 is invalidated before rereading h
        (void)__hip_atomic_load(cnt, __ATOMIC_ACQUIRE, __HIP_MEMORY_SCOPE_AGENT);
    }
}

// ---------------------------------------------------------------------------
// Host launcher
// ---------------------------------------------------------------------------
extern "C" void kernel_launch(void* const* d_in, const int* in_sizes, int n_in,
                              void* d_out, int out_size, void* d_ws, size_t ws_size,
                              hipStream_t stream) {
    (void)in_sizes; (void)n_in; (void)out_size; (void)ws_size;

    const float* x     = (const float*)d_in[0];   // [T, I]
    const float* w_ih  = (const float*)d_in[1];   // [3H, I]
    const float* w_hh  = (const float*)d_in[2];   // [3H, H]
    const float* b     = (const float*)d_in[3];   // [3H]
    const float* b_n   = (const float*)d_in[4];   // [H]
    const float* w_lin = (const float*)d_in[5];   // [O, H]
    const float* b_lin = (const float*)d_in[6];   // [O]
    float* out = (float*)d_out;                   // [T, O]

    // Workspace carve-up (all chunk sizes are multiples of 256 B)
    char* ws = (char*)d_ws;
    size_t off = 0;
    float* igates = (float*)(ws + off);  off += (size_t)DIM_T * 3 * DIM_H * 4; // 192 MB
    __bf16* xb    = (__bf16*)(ws + off); off += (size_t)DIM_T * DIM_I * 2;     // 16 MB
    __bf16* wihb  = (__bf16*)(ws + off); off += (size_t)3 * DIM_H * DIM_I * 2; // 12 MB
    __bf16* wlinb = (__bf16*)(ws + off); off += (size_t)DIM_O * DIM_H * 2;     // 4 MB
    __bf16* hsb   = (__bf16*)(ws + off); off += (size_t)DIM_T * DIM_H * 2;     // 32 MB
    float*  hstate = (float*)(ws + off); off += (size_t)DIM_H * 4;             // 8 KB
    unsigned int* cnt = (unsigned int*)(ws + off);

    // 1) convert GEMM operands to bf16
    cvt_f32_bf16<<<2048, 256, 0, stream>>>(x,     xb,    (size_t)DIM_T * DIM_I);
    cvt_f32_bf16<<<1024, 256, 0, stream>>>(w_ih,  wihb,  (size_t)3 * DIM_H * DIM_I);
    cvt_f32_bf16<<<512,  256, 0, stream>>>(w_lin, wlinb, (size_t)DIM_O * DIM_H);

    // 2) igates = x @ w_ih^T + b   [8192 x 6144], K=1024
    {
        int waves  = (DIM_T / 64) * ((3 * DIM_H) / 64);   // 128 * 96 = 12288
        int blocks = (waves + 7) / 8;
        gemm_bf16_wmma<<<blocks, 256, 0, stream>>>(xb, wihb, b, igates,
                                                   DIM_T, 3 * DIM_H, DIM_I);
    }

    // 3) sequential GRU scan (persistent cooperative kernel)
    init_state<<<1, 256, 0, stream>>>(hstate, cnt);
    gru_scan<<<NB, 256, 0, stream>>>(w_hh, igates, b_n, hstate, hsb, cnt);

    // 4) out = hs @ w_lin^T + b_lin   [8192 x 1024], K=2048
    {
        int waves  = (DIM_T / 64) * (DIM_O / 64);         // 128 * 16 = 2048
        int blocks = (waves + 7) / 8;
        gemm_bf16_wmma<<<blocks, 256, 0, stream>>>(hsb, wlinb, b_lin, out,
                                                   DIM_T, DIM_O, DIM_H);
    }
}
